// SDPriorEncoder_83803401880439
// MI455X (gfx1250) — matrix operand order repaired and checked by reading, not compiled, not run on hardware
//
#include <hip/hip_runtime.h>

// ---------------- problem constants ----------------
#define BB 64
#define EE 500
#define PP 20
#define KTOT (BB*EE)          // 32000 polylines
#define NPTS 5
#define CONVOUT 224
#define HALFC 112
#define KDIM 336              // 112*3 im2col depth
#define KPAD 352              // padded to 11*32
#define NKT 11                // k-tiles of 32
#define NMT 14                // m-tiles of 16 (224/16)
#define COLS (KTOT*NPTS)      // 160000 im2col columns
#define NCT (COLS/16)         // 10000 column tiles
#define FEAT 512
#define TOK 2500              // E*NPTS

// output layout (floats): features | mask | coords
#define OUT_MASK  ((size_t)BB*TOK*FEAT)        // 81,920,000
#define OUT_COORD (OUT_MASK + (size_t)BB*TOK)  // 82,080,000

// workspace layout (units: bf16 halfwords), all 16-half (32B) aligned
#define WS_X1   ((size_t)0)                    // [K][112][8] padded conv1 acts: 28,672,000
#define WS_W2   ((size_t)28672000)             // conv2 A-operand tiles: 78,848
#define WS_L2   ((size_t)(28672000+78848))     // lanes_w2 B-operand: 32,768
#define WS_WD2  ((size_t)(28672000+78848+32768)) // width_w2 B-operand: 32,768

typedef __attribute__((ext_vector_type(16))) __bf16 v16bf;
typedef __attribute__((ext_vector_type(8)))  float  v8f;
typedef __attribute__((ext_vector_type(4)))  unsigned u32x4;
typedef __attribute__((ext_vector_type(8)))  unsigned u32x8;

__device__ __forceinline__ unsigned short f2bf(float f) {
  union { float f; unsigned u; } v; v.f = f;
  unsigned r = v.u + 0x7FFFu + ((v.u >> 16) & 1u);  // round-nearest-even
  return (unsigned short)(r >> 16);
}

// ---------------- kernel 1: weight repack into WMMA operand layouts ----------------
// A-operand 16x32 bf16: lane<16 holds K {0..7,16..23}, lane>=16 holds K {8..15,24..31}
// B-operand 32x16 bf16: lane<16 holds K {0..7,16..23} (N=lane), lane>=16 holds K {8..15,24..31} (N=lane-16)
__global__ void k_prep(const float* __restrict__ conv2_w,
                       const float* __restrict__ lanes_w2,
                       const float* __restrict__ width_w2,
                       unsigned short* __restrict__ ws) {
  int idx = blockIdx.x * 256 + threadIdx.x;
  const int nA = NMT * NKT * 32 * 16;   // 78848
  const int nB = 4 * 16 * 32 * 16;      // 32768
  if (idx < nA) {
    int h = idx & 15, lane = (idx >> 4) & 31;
    int kt = (idx >> 9) % NKT, mt = idx / (NKT * 512);
    int M = mt * 16 + (lane & 15);
    int kk = (lane < 16) ? (h < 8 ? h : h + 8) : (h < 8 ? h + 8 : h + 16);
    int K = kt * 32 + kk;
    float v = 0.f;
    if (K < KDIM) { int i = K / 3, t = K % 3; v = conv2_w[(M * HALFC + i) * 3 + t]; }
    ws[WS_W2 + idx] = f2bf(v);
  } else if (idx < nA + 2 * nB) {
    int j = idx - nA;
    const float* w = (j < nB) ? lanes_w2 : width_w2;
    size_t off = (j < nB) ? WS_L2 : WS_WD2;
    int e = j % nB;
    int h = e & 15, lane = (e >> 4) & 31;
    int nt = (e >> 9) & 15, kt = e >> 13;
    int N = nt * 16 + (lane & 15);
    int K = kt * 32 + ((lane < 16) ? (h < 8 ? h : h + 8) : (h < 8 ? h + 8 : h + 16));
    ws[off + e] = f2bf(w[K * 256 + N]);
  }
}

// ---------------- kernel 2: geometry (resample + PE + conv1) ----------------
// one wave (32 lanes) per polyline; 8 waves / block
__global__ void k_geom(const float* __restrict__ geom,
                       const float* __restrict__ c1w,
                       const float* __restrict__ c1b,
                       unsigned short* __restrict__ ws,
                       float* __restrict__ out) {
  __shared__ float c20[8][PP][2];
  __shared__ float segl[8][PP];
  __shared__ float cum[8][PP];
  __shared__ float c5[8][NPTS][2];
  int w = threadIdx.x >> 5, lane = threadIdx.x & 31;
  int k = blockIdx.x * 8 + w;
  int b = k / EE, e = k % EE;
  const float* g = geom + (size_t)k * PP * 2;
  if (lane < PP) { c20[w][lane][0] = g[lane * 2]; c20[w][lane][1] = g[lane * 2 + 1]; }
  __syncthreads();
  if (lane < PP - 1) {
    float dx = c20[w][lane + 1][0] - c20[w][lane][0];
    float dy = c20[w][lane + 1][1] - c20[w][lane][1];
    segl[w][lane] = sqrtf(dx * dx + dy * dy);
  }
  __syncthreads();
  if (lane == 0) {
    float s = 0.f; cum[w][0] = 0.f;
    for (int i = 0; i < PP - 1; i++) { s += segl[w][i]; cum[w][i + 1] = s; }
  }
  __syncthreads();
  if (lane < NPTS) {
    float total = cum[w][PP - 1];
    float target = total * (0.25f * (float)lane);
    // searchsorted 'left' = count of cum[j] < target, clipped to [1, 19]
    int idx = 1;
    for (int j = 1; j < PP; j++) { if (cum[w][j] < target) idx = j + 1; }
    if (idx > PP - 1) idx = PP - 1;
    float t = (target - cum[w][idx - 1]) / (segl[w][idx - 1] + 1e-8f);
    float px = c20[w][idx - 1][0] + t * (c20[w][idx][0] - c20[w][idx - 1][0]);
    float py = c20[w][idx - 1][1] + t * (c20[w][idx][1] - c20[w][idx - 1][1]);
    if (total < 1e-6f) { px = c20[w][0][0]; py = c20[w][0][1]; }
    float cx = (px + 30.f) / 60.f;   // (p + roi/2) / roi
    float cy = (py + 15.f) / 30.f;
    c5[w][lane][0] = cx; c5[w][lane][1] = cy;
    size_t tk = (size_t)b * TOK + (size_t)e * NPTS + lane;
    out[OUT_COORD + tk * 2]     = cx;
    out[OUT_COORD + tk * 2 + 1] = cy;
    out[OUT_MASK + tk] = 0.f;     // padding mask all-false
  }
  __syncthreads();
  // sinusoidal PE -> feature dims [0,32)
  for (int idx = lane; idx < NPTS * 2 * 8; idx += 32) {
    int f = idx & 7, j = (idx >> 3) & 1, l = idx >> 4;
    float v = c5[w][l][j] * (float)(1 << f) * 3.14159265358979323846f;
    size_t base = ((size_t)b * TOK + (size_t)e * NPTS + l) * FEAT + j * 16 + f;
    out[base]     = sinf(v);
    out[base + 8] = cosf(v);
  }
  // conv1 (2->112, k3, SAME) + ReLU, stored pre-padded: slot p = l+t in [0..7]
  for (int idx = lane; idx < HALFC * NPTS; idx += 32) {
    int i = idx / NPTS, l = idx % NPTS;
    float acc = c1b[i];
    #pragma unroll
    for (int j = 0; j < 2; j++)
      #pragma unroll
      for (int t = 0; t < 3; t++) {
        int xl = l + t - 1;
        float cv = (xl >= 0 && xl < NPTS) ? c5[w][xl][j] : 0.f;
        acc += c1w[(i * 2 + j) * 3 + t] * cv;
      }
    acc = fmaxf(acc, 0.f);
    ws[WS_X1 + ((size_t)k * HALFC + i) * 8 + (l + 1)] = f2bf(acc);
  }
  for (int idx = lane; idx < HALFC * 3; idx += 32) {   // zero pad slots 0,6,7
    int i = idx / 3, pi = idx % 3;
    int p = (pi == 0) ? 0 : (pi == 1 ? 6 : 7);
    ws[WS_X1 + ((size_t)k * HALFC + i) * 8 + p] = 0;
  }
}

// ---------------- kernel 3: conv2 as im2col WMMA GEMM + LayerNorm ----------------
// block = 224 threads (7 waves) per 16-column tile; wave owns m-tiles {w, w+7}.
// The 4 polylines covered by a c-tile are one contiguous 7168B slab of x1pad:
// staged to LDS with a single TDM tensor_load_to_lds (1-D descriptor) from wave 0.
__global__ void k_conv2(const unsigned short* __restrict__ ws,
                        const float* __restrict__ c2b,
                        const float* __restrict__ ng,
                        const float* __restrict__ nb,
                        float* __restrict__ out) {
  __shared__ __align__(32) unsigned short Bp[NKT * 32 * 16];  // 11 B-tiles (swizzled)
  __shared__ __align__(16) unsigned short stage[4 * HALFC * 8]; // raw x1pad rows, 7168 B
  __shared__ float yln[CONVOUT * 16];
  __shared__ float mu[16], rs[16];
  int tid = threadIdx.x;
  int ctile = blockIdx.x;
  int wv = tid >> 5, lane = tid & 31;
  int kbase = (ctile * 16) / NPTS;          // first polyline of this c-tile

  // ---- TDM: DMA the contiguous activation slab (4 polylines) into LDS ----
  if (wv == 0) {
    unsigned ldsoff = (unsigned)(size_t)(&stage[0]);      // low 32 bits = LDS offset
    unsigned long long ga =
        (unsigned long long)(size_t)(ws + WS_X1 + (size_t)kbase * HALFC * 8);
    // D# group 0: count=1 | lds_addr | global_addr[56:0] | type=2
    u32x4 g0;
    g0[0] = 1u;
    g0[1] = ldsoff;
    g0[2] = (unsigned)ga;
    g0[3] = ((unsigned)(ga >> 32) & 0x1FFFFFFu) | 0x80000000u;
    // D# group 1: data_size=3 (8B) | tensor_dim0=896 | tensor_dim1=1 |
    //             tile_dim0=896 | tile_dim1/2=0 | tensor_dim0_stride=896
    u32x8 g1;
    g1[0] = 3u << 16;                 // workgroup_mask=0, data_size=8B
    g1[1] = (896u & 0xFFFFu) << 16;   // tensor_dim0[15:0] at bits 63:48
    g1[2] = 1u << 16;                 // tensor_dim0[31:16]=0, tensor_dim1[15:0]=1
    g1[3] = 896u << 16;               // tensor_dim1[31:16]=0, tile_dim0=896
    g1[4] = 0u;                       // tile_dim1=0 (unused), tile_dim2=0
    g1[5] = 896u;                     // tensor_dim0_stride[31:0]
    g1[6] = 0u;                       // stride bits / tensor_dim1_stride lo
    g1[7] = 0u;
    asm volatile("tensor_load_to_lds %0, %1" :: "s"(g0), "s"(g1) : "memory");
    __builtin_amdgcn_s_wait_tensorcnt(0);
  }
  __syncthreads();

  // ---- build im2col B tiles (swizzled operand layout) from the LDS stage ----
  for (int idx = tid; idx < KPAD * 16; idx += 224) {
    int n = idx / KPAD, r = idx % KPAD;
    int c = ctile * 16 + n, k = c / NPTS, l = c % NPTS;
    unsigned short v = 0;
    if (r < KDIM) {
      int i = r / 3, t = r % 3;
      v = stage[(k - kbase) * (HALFC * 8) + i * 8 + (l + t)];
    }
    int kt = r >> 5, kk = r & 31;
    int lane2 = ((kk >> 3) & 1) ? 16 + n : n;
    Bp[(kt * 32 + lane2) * 16 + ((kk & 7) | ((kk >> 1) & 8))] = v;
  }
  __syncthreads();

  int mt0 = wv, mt1 = wv + 7;
  const unsigned short* W2 = ws + WS_W2;
  __builtin_prefetch(&W2[((size_t)(mt0 * NKT) * 32 + lane) * 16], 0, 0);
  __builtin_prefetch(&W2[((size_t)(mt1 * NKT) * 32 + lane) * 16], 0, 0);
  v8f acc0 = {}; v8f acc1 = {};
  for (int kt = 0; kt < NKT; kt++) {
    v16bf bm = *(const v16bf*)&Bp[(kt * 32 + lane) * 16];
    v16bf a0 = *(const v16bf*)&W2[((size_t)(mt0 * NKT + kt) * 32 + lane) * 16];
    v16bf a1 = *(const v16bf*)&W2[((size_t)(mt1 * NKT + kt) * 32 + lane) * 16];
    acc0 = __builtin_amdgcn_wmma_f32_16x16x32_bf16(false, a0, false, bm, (short)0, acc0, false, false);
    acc1 = __builtin_amdgcn_wmma_f32_16x16x32_bf16(false, a1, false, bm, (short)0, acc1, false, false);
  }
  int nn = lane & 15, mb = (lane < 16) ? 0 : 8;
  #pragma unroll
  for (int j = 0; j < 8; j++) {
    int o0 = mt0 * 16 + mb + j, o1 = mt1 * 16 + mb + j;
    yln[o0 * 16 + nn] = fmaxf(acc0[j] + c2b[o0], 0.f);
    yln[o1 * 16 + nn] = fmaxf(acc1[j] + c2b[o1], 0.f);
  }
  __syncthreads();
  if (tid < 16) {
    float s = 0.f, s2 = 0.f;
    for (int o = 0; o < CONVOUT; o++) { float v = yln[o * 16 + tid]; s += v; s2 += v * v; }
    float m = s / (float)CONVOUT;
    mu[tid] = m; rs[tid] = rsqrtf(s2 / (float)CONVOUT - m * m + 1e-5f);
  }
  __syncthreads();
  for (int idx = tid; idx < CONVOUT * 16; idx += 224) {
    int o = idx >> 4, n = idx & 15;
    int c = ctile * 16 + n, k = c / NPTS, l = c % NPTS, b = k / EE, e = k % EE;
    size_t base = ((size_t)b * TOK + (size_t)e * NPTS + l) * FEAT + 32 + o;
    out[base] = (yln[o * 16 + n] - mu[n]) * rs[n] * ng[o] + nb[o];
  }
}

// ---------------- kernel 4: semantic encoder (MLPs via WMMA) ----------------
// block = 256 threads (8 waves) per 16 entities; wave owns 2 n-tiles (32 channels)
__global__ void k_sem(const int* __restrict__ hw, const int* __restrict__ lanes,
                      const float* __restrict__ width, const int* __restrict__ city,
                      const float* __restrict__ hwE, const float* __restrict__ ctE,
                      const float* __restrict__ l_w1, const float* __restrict__ l_b1,
                      const float* __restrict__ l_b2, const float* __restrict__ l_mk,
                      const float* __restrict__ w_w1, const float* __restrict__ w_b1,
                      const float* __restrict__ w_b2, const float* __restrict__ w_mk,
                      const float* __restrict__ lng, const float* __restrict__ lnb,
                      const unsigned short* __restrict__ ws,
                      float* __restrict__ out) {
  __shared__ __align__(32) unsigned short Al[4 * 32 * 16];
  __shared__ __align__(32) unsigned short Aw[4 * 32 * 16];
  __shared__ float semv[16 * 256];
  __shared__ float lval[16], wval[16];
  __shared__ int lvalid[16], wvalid[16], hwi[16], cti[16];
  __shared__ float mu[16], rs[16];
  int tid = threadIdx.x;
  int k0 = blockIdx.x * 16;
  if (tid < 16) {
    int k = k0 + tid;
    int lv = lanes[k];  lval[tid] = (float)lv; lvalid[tid] = (lv != -1);
    float wf = width[k]; wval[tid] = wf;       wvalid[tid] = (wf != -1.0f);
    hwi[tid] = hw[k]; cti[tid] = city[k];
  }
  __syncthreads();
  // h1 = relu(x*w1 + b1), staged in A-operand layout (16 entities x 128)
  for (int idx = tid; idx < 16 * 128; idx += 256) {
    int m = idx >> 7, f = idx & 127;
    float hL = fmaxf(lval[m] * l_w1[f] + l_b1[f], 0.f);
    float hW = fmaxf(wval[m] * w_w1[f] + w_b1[f], 0.f);
    int kt = f >> 5, kk = f & 31, lane2, h;
    if (kk < 8)       { lane2 = m;      h = kk;      }
    else if (kk < 16) { lane2 = m + 16; h = kk - 8;  }
    else if (kk < 24) { lane2 = m;      h = kk - 8;  }
    else              { lane2 = m + 16; h = kk - 16; }
    int off = (kt * 32 + lane2) * 16 + h;
    Al[off] = f2bf(hL); Aw[off] = f2bf(hW);
  }
  __syncthreads();
  int wv2 = tid >> 5, lane = tid & 31;
  int nt0 = 2 * wv2, nt1 = nt0 + 1;
  const unsigned short* BL = ws + WS_L2;
  const unsigned short* BW = ws + WS_WD2;
  v8f aL0 = {}, aL1 = {}, aW0 = {}, aW1 = {};
  for (int kt = 0; kt < 4; kt++) {
    v16bf al  = *(const v16bf*)&Al[(kt * 32 + lane) * 16];
    v16bf aw  = *(const v16bf*)&Aw[(kt * 32 + lane) * 16];
    v16bf b0l = *(const v16bf*)&BL[((size_t)(kt * 16 + nt0) * 32 + lane) * 16];
    v16bf b1l = *(const v16bf*)&BL[((size_t)(kt * 16 + nt1) * 32 + lane) * 16];
    v16bf b0w = *(const v16bf*)&BW[((size_t)(kt * 16 + nt0) * 32 + lane) * 16];
    v16bf b1w = *(const v16bf*)&BW[((size_t)(kt * 16 + nt1) * 32 + lane) * 16];
    aL0 = __builtin_amdgcn_wmma_f32_16x16x32_bf16(false, al, false, b0l, (short)0, aL0, false, false);
    aL1 = __builtin_amdgcn_wmma_f32_16x16x32_bf16(false, al, false, b1l, (short)0, aL1, false, false);
    aW0 = __builtin_amdgcn_wmma_f32_16x16x32_bf16(false, aw, false, b0w, (short)0, aW0, false, false);
    aW1 = __builtin_amdgcn_wmma_f32_16x16x32_bf16(false, aw, false, b1w, (short)0, aW1, false, false);
  }
  int nn = lane & 15, mb = (lane < 16) ? 0 : 8;
  #pragma unroll
  for (int j = 0; j < 8; j++) {
    int m = mb + j;
    #pragma unroll
    for (int half = 0; half < 2; half++) {
      int n = (half ? nt1 : nt0) * 16 + nn;
      float accL = half ? aL1[j] : aL0[j];
      float accW = half ? aW1[j] : aW0[j];
      float v = hwE[hwi[m] * 256 + n] + ctE[cti[m] * 256 + n]
              + (lvalid[m] ? accL + l_b2[n] : l_mk[n])
              + (wvalid[m] ? accW + w_b2[n] : w_mk[n]);
      semv[m * 256 + n] = v;
    }
  }
  __syncthreads();
  if (tid < 16) {
    float s = 0.f, s2 = 0.f;
    for (int n = 0; n < 256; n++) { float v = semv[tid * 256 + n]; s += v; s2 += v * v; }
    float m = s / 256.f;
    mu[tid] = m; rs[tid] = rsqrtf(s2 / 256.f - m * m + 1e-5f);
  }
  __syncthreads();
  // LayerNorm + broadcast to the entity's 5 tokens, feature dims [256,512)
  for (int idx = tid; idx < 16 * 256; idx += 256) {
    int m = idx >> 8, n = idx & 255;
    float v = (semv[m * 256 + n] - mu[m]) * rs[m] * lng[n] + lnb[n];
    int k = k0 + m, b = k / EE, e = k % EE;
    size_t base = ((size_t)b * TOK + (size_t)e * NPTS) * FEAT + 256 + n;
    #pragma unroll
    for (int l = 0; l < NPTS; l++) out[base + (size_t)l * FEAT] = v;
  }
}

extern "C" void kernel_launch(void* const* d_in, const int* in_sizes, int n_in,
                              void* d_out, int out_size, void* d_ws, size_t ws_size,
                              hipStream_t stream) {
  const float* geometries = (const float*)d_in[0];
  const int*   highway    = (const int*)d_in[1];
  const int*   lanesI     = (const int*)d_in[2];
  const float* widthF     = (const float*)d_in[3];
  const int*   cityI      = (const int*)d_in[4];
  const float* conv1_w = (const float*)d_in[5];
  const float* conv1_b = (const float*)d_in[6];
  const float* conv2_w = (const float*)d_in[7];
  const float* conv2_b = (const float*)d_in[8];
  const float* cng     = (const float*)d_in[9];
  const float* cnb     = (const float*)d_in[10];
  const float* hwE     = (const float*)d_in[11];
  const float* ctE     = (const float*)d_in[12];
  const float* l_w1 = (const float*)d_in[13];
  const float* l_b1 = (const float*)d_in[14];
  const float* l_w2 = (const float*)d_in[15];
  const float* l_b2 = (const float*)d_in[16];
  const float* l_mk = (const float*)d_in[17];
  const float* w_w1 = (const float*)d_in[18];
  const float* w_b1 = (const float*)d_in[19];
  const float* w_w2 = (const float*)d_in[20];
  const float* w_b2 = (const float*)d_in[21];
  const float* w_mk = (const float*)d_in[22];
  const float* lng  = (const float*)d_in[23];
  const float* lnb  = (const float*)d_in[24];
  unsigned short* ws = (unsigned short*)d_ws;
  float* out = (float*)d_out;

  k_prep<<<(144384 + 255) / 256, 256, 0, stream>>>(conv2_w, l_w2, w_w2, ws);
  k_geom<<<KTOT / 8, 256, 0, stream>>>(geometries, conv1_w, conv1_b, ws, out);
  k_conv2<<<NCT, 224, 0, stream>>>(ws, conv2_b, cng, cnb, out);
  k_sem<<<KTOT / 16, 256, 0, stream>>>(highway, lanesI, widthF, cityI, hwE, ctE,
                                       l_w1, l_b1, l_b2, l_mk,
                                       w_w1, w_b1, w_b2, w_mk,
                                       lng, lnb, ws, out);
}
